// Attention_58334245814870
// MI455X (gfx1250) — matrix operand order repaired and checked by reading, not compiled, not run on hardware
//
#include <hip/hip_runtime.h>
#include <hip/hip_bf16.h>

// ---------- types for WMMA ----------
typedef __attribute__((ext_vector_type(16))) __bf16 v16bf;
typedef __attribute__((ext_vector_type(8)))  __bf16 v8bf;
typedef __attribute__((ext_vector_type(8)))  float  v8f;
typedef __attribute__((ext_vector_type(4)))  int    i4v;

// ---------- async global->LDS (CDNA5) with sync fallback ----------
#if defined(__has_builtin)
#if __has_builtin(__builtin_amdgcn_global_load_async_to_lds_b128) && \
    __has_builtin(__builtin_amdgcn_s_wait_asynccnt)
#define USE_ASYNC_LDS 1
#endif
#endif
#ifndef USE_ASYNC_LDS
#define USE_ASYNC_LDS 0
#endif

__device__ inline void copy16B_to_lds(const unsigned short* gsrc, unsigned short* ldst) {
#if USE_ASYNC_LDS
    __builtin_amdgcn_global_load_async_to_lds_b128(
        (__attribute__((address_space(1))) i4v*)(unsigned long long)gsrc,
        (__attribute__((address_space(3))) i4v*)ldst, 0, 0);
#else
    *(uint4*)ldst = *(const uint4*)gsrc;
#endif
}
__device__ inline void async_lds_wait() {
#if USE_ASYNC_LDS
    __builtin_amdgcn_s_wait_asynccnt(0);
#endif
}

// ---------- bf16 helpers (RNE) ----------
__device__ inline unsigned short f2bf_bits(float f) {
    union { float f; unsigned u; } c; c.f = f;
    unsigned u = c.u;
    u += 0x7FFFu + ((u >> 16) & 1u);
    return (unsigned short)(u >> 16);
}

__device__ inline v16bf cat8(v8bf lo, v8bf hi) {
    return __builtin_shufflevector(lo, hi, 0,1,2,3,4,5,6,7,8,9,10,11,12,13,14,15);
}

// A fragment (16x32 bf16, M on lanes; K interleave per ISA 7.12.2):
// low lanes: e0-7 -> K0-7, e8-15 -> K16-23 ; high lanes: +8
__device__ inline v16bf frag_a_lds(const unsigned short* tile, int ld, int lane) {
    int m  = lane & 15;
    int kb = (lane & 16) ? 8 : 0;
    const unsigned short* p = tile + m * ld + kb;
    v8bf lo = *(const v8bf*)p;
    v8bf hi = *(const v8bf*)(p + 16);
    return cat8(lo, hi);
}

// B fragment (32x16 bf16, N on lanes; low lanes: e0-7 -> K0-7, e8-15 -> K16-23 ; high lanes: +8),
// tile stored transposed in LDS: [n][k], k contiguous
__device__ inline v16bf frag_b_lds(const unsigned short* tile, int ld, int lane) {
    int n  = lane & 15;
    int kb = (lane & 16) ? 8 : 0;
    const unsigned short* p = tile + n * ld + kb;
    v8bf lo = *(const v8bf*)p;
    v8bf hi = *(const v8bf*)(p + 16);
    return cat8(lo, hi);
}

__device__ inline v8f wmma_bf16(v16bf a, v16bf b, v8f c) {
    return __builtin_amdgcn_wmma_f32_16x16x32_bf16(false, a, false, b, (short)0, c, false, false);
}

// ---------- problem constants ----------
#define BB   2
#define NN   8000
#define CC   512
#define NH   8
#define HD   64
#define DD20 20
#define NSR  1000

// ---------- convert / transpose kernels ----------
__global__ __launch_bounds__(256) void convert_bf16_kernel(const float* __restrict__ in,
                                                           unsigned short* __restrict__ out, int count) {
    for (int i = blockIdx.x * 256 + threadIdx.x; i < count; i += gridDim.x * 256)
        out[i] = f2bf_bits(in[i]);
}

// in: K x N (row-major) fp32 ; out: N x K (row-major) bf16
__global__ __launch_bounds__(256) void transpose_bf16_kernel(const float* __restrict__ in,
                                                             unsigned short* __restrict__ out,
                                                             int K, int N) {
    int i = blockIdx.x * 256 + threadIdx.x;
    if (i >= N * K) return;
    int n = i / K, k = i % K;
    out[i] = f2bf_bits(in[(size_t)k * N + n]);
}

// ---------- generic bf16 GEMM: out = A(MxK) * Bt(NxK)^T + bias ----------
// MODE 0: q proj   -> outK bf16 row-major (MxN)
// MODE 1: kv proj  -> split k/v into (B,h,Nsr,hd) bf16, plus v fp32 (rows x 512)
// MODE 2: out proj -> outF fp32 row-major (MxN)
template <int MODE>
__global__ __launch_bounds__(256) void gemm_bf16_kernel(
    const unsigned short* __restrict__ A,
    const unsigned short* __restrict__ Bt,
    const float* __restrict__ bias,
    float* __restrict__ outF,
    unsigned short* __restrict__ outK,
    unsigned short* __restrict__ outV,
    float* __restrict__ outVF,
    int M, int N, int K) {
    __shared__ __align__(16) unsigned short lA[64 * 40];
    __shared__ __align__(16) unsigned short lB[128 * 40];

    int tid = threadIdx.x;
    int lane = tid & 31, wave = tid >> 5;
    int wm = wave >> 2, wn = wave & 3;
    int m0 = blockIdx.x * 64;
    int n0 = blockIdx.y * 128;

    v8f acc[2][2];
    for (int i = 0; i < 2; ++i)
        for (int j = 0; j < 2; ++j)
            for (int e = 0; e < 8; ++e) acc[i][j][e] = 0.f;

    int ar = tid >> 2, as = tid & 3;   // A tile: 64 rows x 4 segments of 8 bf16
    int bn = tid >> 1, bseg = tid & 1; // B tile: 128 rows x 2 segments of 16 bf16
    int gra = m0 + ar;
    if (gra >= M) gra = M - 1;         // clamp: out-of-range rows never stored

    for (int k0 = 0; k0 < K; k0 += 32) {
        const unsigned short* asrc = A + (size_t)gra * K + k0 + as * 8;
        copy16B_to_lds(asrc, lA + ar * 40 + as * 8);

        const unsigned short* bsrc = Bt + (size_t)(n0 + bn) * K + k0 + bseg * 16;
        copy16B_to_lds(bsrc,     lB + bn * 40 + bseg * 16);
        copy16B_to_lds(bsrc + 8, lB + bn * 40 + bseg * 16 + 8);

        async_lds_wait();
        __syncthreads();

        v16bf af[2], bfm[2];
        for (int i = 0; i < 2; ++i) af[i]  = frag_a_lds(lA + (wm * 32 + i * 16) * 40, 40, lane);
        for (int j = 0; j < 2; ++j) bfm[j] = frag_b_lds(lB + (wn * 32 + j * 16) * 40, 40, lane);
        for (int i = 0; i < 2; ++i)
            for (int j = 0; j < 2; ++j)
                acc[i][j] = wmma_bf16(af[i], bfm[j], acc[i][j]);
        __syncthreads();
    }

    int rh = (lane >> 4) & 1;
    for (int i = 0; i < 2; ++i) {
        for (int j = 0; j < 2; ++j) {
            int col = n0 + wn * 32 + j * 16 + (lane & 15);
            float bcol = bias[col];
            for (int e = 0; e < 8; ++e) {
                int gr = m0 + wm * 32 + i * 16 + e + rh * 8;
                if (gr >= M) continue;
                float val = acc[i][j][e] + bcol;
                if (MODE == 0) {
                    outK[(size_t)gr * N + col] = f2bf_bits(val);
                } else if (MODE == 2) {
                    outF[(size_t)gr * N + col] = val;
                } else {
                    int b = gr / NSR, nr = gr % NSR;
                    int t = col >> 9, hh = (col >> 6) & 7, d = col & 63;
                    size_t di = ((size_t)(b * NH + hh) * NSR + nr) * HD + d;
                    if (t == 0) {
                        outK[di] = f2bf_bits(val);
                    } else {
                        outV[di] = f2bf_bits(val);
                        outVF[(size_t)gr * CC + (col - CC)] = val;
                    }
                }
            }
        }
    }
}

// ---------- depthwise conv3d k=3 s=2 p=1: x(B,N,C) -> xs(B,Nsr,C) + bias ----------
__global__ __launch_bounds__(256) void srconv_kernel(const float* __restrict__ x,
                                                     const float* __restrict__ w,
                                                     const float* __restrict__ bias,
                                                     float* __restrict__ xs) {
    int idx = blockIdx.x;           // b*1000 + spatial
    int b = idx / NSR, sp = idx % NSR;
    int od = sp / 100, oh = (sp / 10) % 10, ow = sp % 10;
    for (int c = threadIdx.x; c < CC; c += 256) {
        float accv = bias[c];
        for (int kd = 0; kd < 3; ++kd) {
            int id = od * 2 - 1 + kd;
            if (id < 0 || id >= DD20) continue;
            for (int kh = 0; kh < 3; ++kh) {
                int ih = oh * 2 - 1 + kh;
                if (ih < 0 || ih >= DD20) continue;
                for (int kw = 0; kw < 3; ++kw) {
                    int iw = ow * 2 - 1 + kw;
                    if (iw < 0 || iw >= DD20) continue;
                    accv += x[((size_t)b * NN + id * 400 + ih * 20 + iw) * CC + c] *
                            w[c * 27 + kd * 9 + kh * 3 + kw];
                }
            }
        }
        xs[(size_t)idx * CC + c] = accv;
    }
}

// ---------- rowwise layernorm (C=512) -> bf16 ----------
__global__ __launch_bounds__(256) void ln_rows_kernel(const float* __restrict__ in,
                                                      const float* __restrict__ g,
                                                      const float* __restrict__ be,
                                                      unsigned short* __restrict__ outb) {
    __shared__ float r1[256], r2[256];
    int row = blockIdx.x, tid = threadIdx.x;
    const float* p = in + (size_t)row * CC;
    float a = p[tid], b = p[tid + 256];
    r1[tid] = a + b;
    r2[tid] = a * a + b * b;
    __syncthreads();
    for (int s = 128; s > 0; s >>= 1) {
        if (tid < s) { r1[tid] += r1[tid + s]; r2[tid] += r2[tid + s]; }
        __syncthreads();
    }
    float mean = r1[0] * (1.f / CC);
    float var  = r2[0] * (1.f / CC) - mean * mean;
    float inv  = rsqrtf(var + 1e-6f);
    outb[(size_t)row * CC + tid]       = f2bf_bits((a - mean) * inv * g[tid] + be[tid]);
    outb[(size_t)row * CC + tid + 256] = f2bf_bits((b - mean) * inv * g[tid + 256] + be[tid + 256]);
}

// ---------- flash attention: o(b,n,c) = softmax(q k^T * 0.125) v ----------
__global__ __launch_bounds__(128) void attn_kernel(const unsigned short* __restrict__ qb,
                                                   const unsigned short* __restrict__ kmat,
                                                   const unsigned short* __restrict__ vmat,
                                                   float* __restrict__ o) {
    __shared__ __align__(16) unsigned short sK[32 * 72];   // [key][64+8]
    __shared__ __align__(16) unsigned short sVt[64 * 40];  // [d][32+8]
    __shared__ __align__(16) float sS[4][16 * 36];         // per-wave S f32
    __shared__ __align__(16) unsigned short sP[4][16 * 40];// per-wave P bf16
    __shared__ float sRow[4][16];
    __shared__ float sSum[4][16];

    int tid = threadIdx.x;
    int lane = tid & 31, wave = tid >> 5;
    int qt = blockIdx.x % 125;
    int bh = blockIdx.x / 125;
    int b = bh >> 3, hh = bh & 7;
    int q0 = qt * 64;

    // q fragments (2 K-chunks of 32), loaded straight from global bf16
    int kbq = (lane & 16) ? 8 : 0;
    const unsigned short* qrow =
        qb + ((size_t)(b * NN + q0 + wave * 16 + (lane & 15))) * CC + hh * HD;
    v16bf aq[2];
    for (int c = 0; c < 2; ++c) {
        v8bf lo = *(const v8bf*)(qrow + c * 32 + kbq);
        v8bf hi = *(const v8bf*)(qrow + c * 32 + kbq + 16);
        aq[c] = cat8(lo, hi);
    }

    v8f acc[4];
    for (int j = 0; j < 4; ++j)
        for (int e = 0; e < 8; ++e) acc[j][e] = 0.f;
    float mrun = -1e30f, lsum = 0.f;

    int r = lane & 15;
    int half = (lane >> 4) & 1;
    const size_t kvbase = (size_t)bh * NSR * HD;

    int key = tid >> 2, seg = tid & 3;

    for (int kt = 0; kt < 32; ++kt) {
        int j0 = kt * 32;
        {   // cooperative K/V tile staging (128 threads; 32 keys x 4 segs)
            int gk = j0 + key;
            if (gk >= NSR) gk = NSR - 1;   // clamp: masked keys get p = 0 anyway
            const unsigned short* ksrc = kmat + kvbase + (size_t)gk * HD + seg * 16;
            copy16B_to_lds(ksrc,     sK + key * 72 + seg * 16);
            copy16B_to_lds(ksrc + 8, sK + key * 72 + seg * 16 + 8);
            const unsigned short* vsrc = vmat + kvbase + (size_t)gk * HD + seg * 16;
            for (int e = 0; e < 16; ++e) sVt[(seg * 16 + e) * 40 + key] = vsrc[e];
            // prefetch next key tile into cache
            int gk2 = j0 + 32 + key;
            if (gk2 < NSR) {
                __builtin_prefetch((const void*)(kmat + kvbase + (size_t)gk2 * HD + seg * 16), 0, 0);
                __builtin_prefetch((const void*)(vmat + kvbase + (size_t)gk2 * HD + seg * 16), 0, 0);
            }
        }
        async_lds_wait();
        __syncthreads();

        // S = q * k^T  (16 rows x 32 keys)
        v8f s[2];
        for (int ks = 0; ks < 2; ++ks)
            for (int e = 0; e < 8; ++e) s[ks][e] = 0.f;
        for (int kc = 0; kc < 2; ++kc)
            for (int ks = 0; ks < 2; ++ks) {
                v16bf bk = frag_b_lds(sK + ks * 16 * 72 + kc * 32, 72, lane);
                s[ks] = wmma_bf16(aq[kc], bk, s[ks]);
            }
        for (int ks = 0; ks < 2; ++ks) {
            int colw = ks * 16 + (lane & 15);
            for (int e = 0; e < 8; ++e)
                sS[wave][(e + half * 8) * 36 + colw] = s[ks][e];
        }
        __syncthreads();

        // online softmax: lane owns row r, keys [half*16, half*16+16)
        float sv[16];
        float tmax = -1e30f;
        for (int t = 0; t < 16; ++t) {
            int jl = half * 16 + t;
            float sval = sS[wave][r * 36 + jl] * 0.125f;
            if (j0 + jl >= NSR) sval = -1e30f;
            sv[t] = sval;
            tmax = fmaxf(tmax, sval);
        }
        tmax = fmaxf(tmax, __shfl_xor(tmax, 16, 32));
        float mnew = fmaxf(mrun, tmax);
        float scl = __expf(mrun - mnew);
        float psum = 0.f;
        for (int t = 0; t < 16; ++t) {
            int jl = half * 16 + t;
            float pv = __expf(sv[t] - mnew);
            psum += pv;
            sP[wave][r * 40 + jl] = f2bf_bits(pv);
        }
        psum += __shfl_xor(psum, 16, 32);
        lsum = lsum * scl + psum;
        mrun = mnew;
        if (half == 0) sRow[wave][r] = scl;
        __syncthreads();

        // rescale accumulators, then O += P * V
        float rs[8];
        for (int e = 0; e < 8; ++e) rs[e] = sRow[wave][e + half * 8];
        v16bf pa = frag_a_lds(sP[wave], 40, lane);
        for (int j = 0; j < 4; ++j) {
            v16bf vbf = frag_b_lds(sVt + j * 16 * 40, 40, lane);
            for (int e = 0; e < 8; ++e) acc[j][e] *= rs[e];
            acc[j] = wmma_bf16(pa, vbf, acc[j]);
        }
        __syncthreads();
    }

    if (half == 0) sSum[wave][r] = lsum;
    __syncthreads();
    float inv[8];
    for (int e = 0; e < 8; ++e) inv[e] = 1.f / sSum[wave][e + half * 8];

    for (int j = 0; j < 4; ++j) {
        int col = hh * HD + j * 16 + (lane & 15);
        for (int e = 0; e < 8; ++e) {
            int gr = b * NN + q0 + wave * 16 + e + half * 8;
            o[(size_t)gr * CC + col] = acc[j][e] * inv[e];
        }
    }
}

// ---------- trilinear upsample of v + LN + add attention output -> bf16 ----------
__global__ __launch_bounds__(256) void upsample_ln_add_kernel(const float* __restrict__ vf,
                                                              const float* __restrict__ o,
                                                              const float* __restrict__ g,
                                                              const float* __restrict__ be,
                                                              unsigned short* __restrict__ yb) {
    __shared__ float r1[256], r2[256];
    int idx = blockIdx.x, tid = threadIdx.x;   // idx = b*8000 + n
    int b = idx / NN, n = idx % NN;
    int d = n / 400, h2 = (n / 20) % 20, w2 = n % 20;

    auto axis = [](int i, int& i0, int& i1, float& t) {
        float cs = i * 0.5f - 0.25f;
        float fl = floorf(cs);
        t = cs - fl;
        int ii = (int)fl;
        i0 = ii < 0 ? 0 : (ii > 9 ? 9 : ii);
        int jj = ii + 1;
        i1 = jj < 0 ? 0 : (jj > 9 ? 9 : jj);
    };
    int d0, d1, h0, h1, w0, w1;
    float td, th, tw;
    axis(d, d0, d1, td);
    axis(h2, h0, h1, th);
    axis(w2, w0, w1, tw);

    const float* base = vf + (size_t)b * NSR * CC;
    float idv[2];
    for (int q = 0; q < 2; ++q) {
        int c = tid + q * 256;
#define VAT(dd, hh, ww) base[(size_t)((dd)*100 + (hh)*10 + (ww)) * CC + c]
        float v000 = VAT(d0, h0, w0), v001 = VAT(d0, h0, w1);
        float v010 = VAT(d0, h1, w0), v011 = VAT(d0, h1, w1);
        float v100 = VAT(d1, h0, w0), v101 = VAT(d1, h0, w1);
        float v110 = VAT(d1, h1, w0), v111 = VAT(d1, h1, w1);
#undef VAT
        float a00 = v000 + (v001 - v000) * tw;
        float a01 = v010 + (v011 - v010) * tw;
        float a10 = v100 + (v101 - v100) * tw;
        float a11 = v110 + (v111 - v110) * tw;
        float b0 = a00 + (a01 - a00) * th;
        float b1 = a10 + (a11 - a10) * th;
        idv[q] = b0 + (b1 - b0) * td;
    }
    r1[tid] = idv[0] + idv[1];
    r2[tid] = idv[0] * idv[0] + idv[1] * idv[1];
    __syncthreads();
    for (int s = 128; s > 0; s >>= 1) {
        if (tid < s) { r1[tid] += r1[tid + s]; r2[tid] += r2[tid + s]; }
        __syncthreads();
    }
    float mean = r1[0] * (1.f / CC);
    float var  = r2[0] * (1.f / CC) - mean * mean;
    float inv  = rsqrtf(var + 1e-6f);
    for (int q = 0; q < 2; ++q) {
        int c = tid + q * 256;
        float ln = (idv[q] - mean) * inv * g[c] + be[c];
        yb[(size_t)idx * CC + c] = f2bf_bits(o[(size_t)idx * CC + c] + ln);
    }
}

// ---------- host driver ----------
extern "C" void kernel_launch(void* const* d_in, const int* in_sizes, int n_in,
                              void* d_out, int out_size, void* d_ws, size_t ws_size,
                              hipStream_t stream) {
    const float* x    = (const float*)d_in[0];
    const float* Wq   = (const float*)d_in[1];
    const float* bq   = (const float*)d_in[2];
    const float* Wkv  = (const float*)d_in[3];
    const float* bkv  = (const float*)d_in[4];
    const float* srw  = (const float*)d_in[5];
    const float* srb  = (const float*)d_in[6];
    const float* srg  = (const float*)d_in[7];
    const float* srbe = (const float*)d_in[8];
    const float* upg  = (const float*)d_in[9];
    const float* upbe = (const float*)d_in[10];
    const float* Wp   = (const float*)d_in[11];
    const float* bp   = (const float*)d_in[12];
    float* out = (float*)d_out;
    (void)in_sizes; (void)n_in; (void)out_size; (void)ws_size;

    char* ws = (char*)d_ws;
    size_t off = 0;
    auto alloc = [&](size_t bytes) -> char* {
        char* p = ws + off;
        off = (off + bytes + 255) & ~(size_t)255;
        return p;
    };

    unsigned short* x_bf  = (unsigned short*)alloc((size_t)NN * BB * CC * 2);
    unsigned short* Wq_t  = (unsigned short*)alloc((size_t)CC * CC * 2);
    unsigned short* Wkv_t = (unsigned short*)alloc((size_t)CC * 2 * CC * 2);
    unsigned short* Wp_t  = (unsigned short*)alloc((size_t)CC * CC * 2);
    unsigned short* q_bf  = (unsigned short*)alloc((size_t)NN * BB * CC * 2);
    float* xs             = (float*)alloc((size_t)BB * NSR * CC * 4);
    unsigned short* xr_bf = (unsigned short*)alloc((size_t)BB * NSR * CC * 2);
    unsigned short* k_bf  = (unsigned short*)alloc((size_t)BB * NH * NSR * HD * 2);
    unsigned short* v_bf  = (unsigned short*)alloc((size_t)BB * NH * NSR * HD * 2);
    float* v_f32          = (float*)alloc((size_t)BB * NSR * CC * 4);
    float* o_f32          = (float*)alloc((size_t)NN * BB * CC * 4);
    unsigned short* y_bf  = (unsigned short*)alloc((size_t)NN * BB * CC * 2);

    const int Mx = BB * NN;    // 16000
    const int Mr = BB * NSR;   // 2000

    convert_bf16_kernel<<<4096, 256, 0, stream>>>(x, x_bf, Mx * CC);
    transpose_bf16_kernel<<<(CC * CC + 255) / 256, 256, 0, stream>>>(Wq, Wq_t, CC, CC);
    transpose_bf16_kernel<<<(CC * 2 * CC + 255) / 256, 256, 0, stream>>>(Wkv, Wkv_t, CC, 2 * CC);
    transpose_bf16_kernel<<<(CC * CC + 255) / 256, 256, 0, stream>>>(Wp, Wp_t, CC, CC);

    dim3 gq((Mx + 63) / 64, CC / 128);
    gemm_bf16_kernel<0><<<gq, 256, 0, stream>>>(x_bf, Wq_t, bq, nullptr, q_bf, nullptr, nullptr,
                                                Mx, CC, CC);

    srconv_kernel<<<Mr, 256, 0, stream>>>(x, srw, srb, xs);
    ln_rows_kernel<<<Mr, 256, 0, stream>>>(xs, srg, srbe, xr_bf);

    dim3 gkv((Mr + 63) / 64, (2 * CC) / 128);
    gemm_bf16_kernel<1><<<gkv, 256, 0, stream>>>(xr_bf, Wkv_t, bkv, nullptr, k_bf, v_bf, v_f32,
                                                 Mr, 2 * CC, CC);

    attn_kernel<<<BB * NH * (NN / 64), 128, 0, stream>>>(q_bf, k_bf, v_bf, o_f32);

    upsample_ln_add_kernel<<<Mx, 256, 0, stream>>>(v_f32, o_f32, upg, upbe, y_bf);

    dim3 gp((Mx + 63) / 64, CC / 128);
    gemm_bf16_kernel<2><<<gp, 256, 0, stream>>>(y_bf, Wp_t, bp, out, nullptr, nullptr, nullptr,
                                                Mx, CC, CC);
}
